// VisualSpeechTextQFormer_42863773614226
// MI455X (gfx1250) — matrix operand order repaired and checked by reading, not compiled, not run on hardware
//
#include <hip/hip_runtime.h>
#include <hip/hip_bf16.h>

// ---------------------------------------------------------------------------
// VisualSpeechTextQFormer forward for gfx1250 (MI455X, wave32, WMMA).
//
// All heavy GEMMs run on v_wmma_f32_16x16x32_bf16 (bf16 in, f32 acc); each
// wave produces a 16x64 C strip (4 WMMAs reusing one A fragment) so the inner
// loop does 4 WMMAs per 10 b128 loads. Weights live in the 192MB L2 after the
// first pass, so HBM traffic ~= one pass over bf16 activations (model is
// HBM-bound: ~0.6 TFLOP vs ~0.6GB traffic at 23.3 TB/s). bias/GELU/residual
// are fused into GEMM epilogues; LayerNorm emits bf16 for the next GEMM.
// The encoder mask makes sequence rows attend only to themselves and query
// rows attend to <=39 keys, so attention is cheap wave32 VALU work
// (shfl_xor reductions, lane-distributed score registers).
// ---------------------------------------------------------------------------

typedef __bf16 bf16;
typedef __attribute__((ext_vector_type(16))) __bf16 v16bf;
typedef __attribute__((ext_vector_type(8)))  __bf16 v8bf;
typedef __attribute__((ext_vector_type(8)))  float  v8f;

#define BQ   8       // batch
#define TT   1024    // frames
#define DD   1024    // d_model
#define NQ   32      // queries
#define NH   8       // heads
#define HD   128     // head dim
#define SEQ  1056    // NQ + TT
#define LL   256     // text length
#define TD   768     // text dim

__device__ inline float gelu_f(float x) {
  return 0.5f * x * (1.0f + erff(x * 0.7071067811865476f));
}
__device__ inline float wave_sum(float v) {
#pragma unroll
  for (int o = 16; o > 0; o >>= 1) v += __shfl_xor(v, o);
  return v;
}
__device__ inline float wave_max(float v) {
#pragma unroll
  for (int o = 16; o > 0; o >>= 1) v = fmaxf(v, __shfl_xor(v, o));
  return v;
}

// ------------------------------ cvt f32 -> bf16 ----------------------------
__global__ void cvt_bf16_k(const float* __restrict__ x, bf16* __restrict__ o, long n) {
  long i = (long)blockIdx.x * 256 + threadIdx.x;
  if (i < n) o[i] = (bf16)x[i];
}

// ------------------------- grouped conv1 + GELU ----------------------------
// x:[B,T,D] f32, w:[D,128,3], out h1 bf16 [B*T, D]
__global__ void conv1_gelu_k(const float* __restrict__ x, const float* __restrict__ w,
                             const float* __restrict__ bias, bf16* __restrict__ h1) {
  int tid = threadIdx.x;
  int co  = ((blockIdx.x & 3) << 8) + tid;       // 0..1023
  int bt  = blockIdx.x >> 2;                     // 0..8191
  int t = bt & (TT - 1);
  int b = bt >> 10;
  int g = co >> 7;
  const float* wp = w + (size_t)co * 384;        // [ci][tap]
  float acc = bias[co];
#pragma unroll
  for (int tap = 0; tap < 3; ++tap) {
    int tt = t + tap - 1;
    if (tt < 0 || tt >= TT) continue;
    const float* xp = x + ((size_t)(b * TT + tt)) * DD + g * 128;
    for (int ci = 0; ci < 128; ++ci) acc += wp[ci * 3 + tap] * xp[ci];
  }
  h1[(size_t)bt * DD + co] = (bf16)gelu_f(acc);
}

// --------------------------- im2col for conv2 ------------------------------
// h1 bf16 [B*T, D] -> A2 bf16 [8192, 3072]  (k = c*3 + tap, matching conv2_w)
__global__ void im2col3_k(const bf16* __restrict__ h1, bf16* __restrict__ A2) {
  size_t idx = (size_t)blockIdx.x * 256 + threadIdx.x;   // 8192*3072 total
  int kk = (int)(idx % 3072);
  size_t m = idx / 3072;
  int c = kk / 3, tap = kk % 3;
  int t = (int)(m & (TT - 1));
  size_t b = m >> 10;
  int tt = t + tap - 1;
  bf16 v = (bf16)0.0f;
  if (tt >= 0 && tt < TT) v = h1[((b << 10) + tt) * DD + c];
  A2[idx] = v;
}

// -------------------- seq assembly: queries + h2 + pos ---------------------
__global__ void build_seq_k(const float* __restrict__ h2, const float* __restrict__ qt,
                            float* __restrict__ hcat) {
  size_t idx = (size_t)blockIdx.x * 256 + threadIdx.x;   // B*SEQ*D total
  int c = (int)(idx & (DD - 1));
  size_t row = idx >> 10;
  int r = (int)(row % SEQ);
  size_t b = row / SEQ;
  float v;
  if (r < NQ) v = qt[(size_t)r * DD + c];
  else {
    int t = r - NQ;
    v = h2[((b << 10) + t) * DD + c] + 0.1f * ((float)t / (float)TT);
  }
  hcat[idx] = v;
}

// ------------------------- LayerNorm -> bf16 -------------------------------
// src row = (row/inner)*outer + (row%inner); identity: inner=rows, outer=0
__global__ void layernorm_bf16_k(const float* __restrict__ x, const float* __restrict__ gam,
                                 const float* __restrict__ bet, bf16* __restrict__ o,
                                 int inner, int outer) {
  int row = blockIdx.x;
  int src = (row / inner) * outer + (row % inner);
  int tid = threadIdx.x;                       // 256 threads, D = 1024
  const float4* xr = (const float4*)(x + (size_t)src * DD);
  float4 v = xr[tid];
  __shared__ float red[256];
  red[tid] = v.x + v.y + v.z + v.w;
  __syncthreads();
  for (int st = 128; st > 0; st >>= 1) { if (tid < st) red[tid] += red[tid + st]; __syncthreads(); }
  float mean = red[0] * (1.0f / DD);
  __syncthreads();
  float dx = v.x - mean, dy = v.y - mean, dz = v.z - mean, dw = v.w - mean;
  red[tid] = dx * dx + dy * dy + dz * dz + dw * dw;
  __syncthreads();
  for (int st = 128; st > 0; st >>= 1) { if (tid < st) red[tid] += red[tid + st]; __syncthreads(); }
  float rstd = rsqrtf(red[0] * (1.0f / DD) + 1e-5f);
  int c = tid * 4;
  bf16* orow = o + (size_t)row * DD + c;
  orow[0] = (bf16)(dx * rstd * gam[c + 0] + bet[c + 0]);
  orow[1] = (bf16)(dy * rstd * gam[c + 1] + bet[c + 1]);
  orow[2] = (bf16)(dz * rstd * gam[c + 2] + bet[c + 2]);
  orow[3] = (bf16)(dw * rstd * gam[c + 3] + bet[c + 3]);
}

// ------------------------------- WMMA GEMM ---------------------------------
// C[M,N] = act(A[M,K](bf16) @ W[N,K](bf16)^T + bias) (+ resid); A,W row-major
// along K. Each wave: 16x64 C strip = 4 x v_wmma_f32_16x16x32_bf16 per K-step
// of 32, sharing one A fragment (4x A reuse). Block = 128 threads = 4 waves
// covering a 16x256 strip of C.
__global__ void gemm_wmma_k(const bf16* __restrict__ A, const bf16* __restrict__ W,
                            const float* __restrict__ bias, const float* __restrict__ resid,
                            float* __restrict__ outF, bf16* __restrict__ outB,
                            int M, int N, int K, int act) {
  const int lane = threadIdx.x & 31;
  const int wave = threadIdx.x >> 5;
  const int nb = blockIdx.x * 256 + wave * 64;   // wave strip: cols nb..nb+63
  const int mb = blockIdx.y * 16;
  const int half = lane >> 4;        // 0: lanes 0-15, 1: lanes 16-31
  const int l16  = lane & 15;
  const bf16* Ap = A + (size_t)(mb + l16) * K;   // A row for this lane (M stripe)
  const bf16* Wp0 = W + (size_t)(nb +      l16) * K;
  const bf16* Wp1 = W + (size_t)(nb + 16 + l16) * K;
  const bf16* Wp2 = W + (size_t)(nb + 32 + l16) * K;
  const bf16* Wp3 = W + (size_t)(nb + 48 + l16) * K;
  v8f acc0 = {}, acc1 = {}, acc2 = {}, acc3 = {};
  for (int k = 0; k < K; k += 32) {
    __builtin_prefetch(Ap + k + 64, 0, 3);
    // A 16x32 bf16 layout: lane half selects contiguous K range half*16..+15
    v8bf a0 = *(const v8bf*)(Ap + k + half * 16);
    v8bf a1 = *(const v8bf*)(Ap + k + half * 16 + 8);
    v16bf a;
#pragma unroll
    for (int i = 0; i < 8; ++i) { a[i] = a0[i]; a[8 + i] = a1[i]; }
    // B 32x16 bf16 layout: lane half selects contiguous K range half*16..+15
    const int kb = k + half * 16;
    {
      v8bf b0 = *(const v8bf*)(Wp0 + kb);
      v8bf b1 = *(const v8bf*)(Wp0 + kb + 8);
      v16bf bm;
#pragma unroll
      for (int i = 0; i < 8; ++i) { bm[i] = b0[i]; bm[8 + i] = b1[i]; }
      acc0 = __builtin_amdgcn_wmma_f32_16x16x32_bf16(false, a, false, bm, (short)0, acc0,
                                                     false, false);
    }
    {
      v8bf b0 = *(const v8bf*)(Wp1 + kb);
      v8bf b1 = *(const v8bf*)(Wp1 + kb + 8);
      v16bf bm;
#pragma unroll
      for (int i = 0; i < 8; ++i) { bm[i] = b0[i]; bm[8 + i] = b1[i]; }
      acc1 = __builtin_amdgcn_wmma_f32_16x16x32_bf16(false, a, false, bm, (short)0, acc1,
                                                     false, false);
    }
    {
      v8bf b0 = *(const v8bf*)(Wp2 + kb);
      v8bf b1 = *(const v8bf*)(Wp2 + kb + 8);
      v16bf bm;
#pragma unroll
      for (int i = 0; i < 8; ++i) { bm[i] = b0[i]; bm[8 + i] = b1[i]; }
      acc2 = __builtin_amdgcn_wmma_f32_16x16x32_bf16(false, a, false, bm, (short)0, acc2,
                                                     false, false);
    }
    {
      v8bf b0 = *(const v8bf*)(Wp3 + kb);
      v8bf b1 = *(const v8bf*)(Wp3 + kb + 8);
      v16bf bm;
#pragma unroll
      for (int i = 0; i < 8; ++i) { bm[i] = b0[i]; bm[8 + i] = b1[i]; }
      acc3 = __builtin_amdgcn_wmma_f32_16x16x32_bf16(false, a, false, bm, (short)0, acc3,
                                                     false, false);
    }
  }
  // epilogue: C lane mapping: col = tile_n + lane%16, row = mb + (lane/16)*8 + r
  v8f accs[4] = {acc0, acc1, acc2, acc3};
#pragma unroll
  for (int s = 0; s < 4; ++s) {
    const int col = nb + s * 16 + l16;
    const float bval = bias ? bias[col] : 0.0f;
#pragma unroll
    for (int r = 0; r < 8; ++r) {
      int row = mb + half * 8 + r;
      float v = accs[s][r] + bval;
      if (act == 1) v = gelu_f(v);
      if (resid) v += resid[(size_t)row * N + col];
      if (outF) outF[(size_t)row * N + col] = v;
      else      outB[(size_t)row * N + col] = (bf16)v;
    }
  }
}

// ------------- encoder attention: sequence rows attend to self -------------
__global__ void copy_v_seq_k(const float* __restrict__ qkv, bf16* __restrict__ ao) {
  size_t idx = (size_t)blockIdx.x * 256 + threadIdx.x;   // B*SEQ*D
  int c = (int)(idx & (DD - 1));
  size_t row = idx >> 10;
  if ((int)(row % SEQ) < NQ) return;                     // query rows done elsewhere
  ao[row * DD + c] = (bf16)qkv[row * 3072 + 2048 + c];
}

// ---- encoder query attention: one wave per (b,h,qi), <=39 masked keys -----
__global__ void enc_qattn_k(const float* __restrict__ qkv, bf16* __restrict__ ao) {
  int wid  = (int)((blockIdx.x * (size_t)blockDim.x + threadIdx.x) >> 5);
  int lane = threadIdx.x & 31;
  if (wid >= BQ * NH * NQ) return;
  int qi = wid & 31, h = (wid >> 5) & 7, b = wid >> 8;
  const float scale = 0.08838834764831845f;              // 1/sqrt(128)
  size_t base = (size_t)b * SEQ * 3072;
  const float* qp = qkv + base + (size_t)qi * 3072 + h * HD + lane * 4;
  float q0 = qp[0], q1 = qp[1], q2 = qp[2], q3 = qp[3];
  int rs = qi * 32 - 3; if (rs < 0) rs = 0;
  int re = qi * 32 + 35; if (re > TT) re = TT;
  int ncols = 1 + (re - rs);                             // self + window
  float s0 = -1e30f, s1 = -1e30f;                        // scores distributed over lanes
  for (int idx = 0; idx < ncols; ++idx) {
    int sj = (idx == 0) ? qi : (NQ + rs + idx - 1);
    const float* kp = qkv + base + (size_t)sj * 3072 + 1024 + h * HD + lane * 4;
    float p = q0 * kp[0] + q1 * kp[1] + q2 * kp[2] + q3 * kp[3];
    p = wave_sum(p) * scale;
    if (idx < 32) { if (lane == idx)      s0 = p; }
    else          { if (lane == idx - 32) s1 = p; }
  }
  float m = wave_max(fmaxf(s0, s1));
  float w0 = (s0 > -1e29f) ? __expf(s0 - m) : 0.0f;
  float w1 = (s1 > -1e29f) ? __expf(s1 - m) : 0.0f;
  float l = wave_sum(w0 + w1);
  float o0 = 0, o1 = 0, o2 = 0, o3 = 0;
  for (int idx = 0; idx < ncols; ++idx) {
    int sj = (idx == 0) ? qi : (NQ + rs + idx - 1);
    float w = __shfl(idx < 32 ? w0 : w1, idx & 31);
    const float* vp = qkv + base + (size_t)sj * 3072 + 2048 + h * HD + lane * 4;
    o0 += w * vp[0]; o1 += w * vp[1]; o2 += w * vp[2]; o3 += w * vp[3];
  }
  float inv = 1.0f / l;
  bf16* op = ao + ((size_t)b * SEQ + qi) * DD + h * HD + lane * 4;
  op[0] = (bf16)(o0 * inv); op[1] = (bf16)(o1 * inv);
  op[2] = (bf16)(o2 * inv); op[3] = (bf16)(o3 * inv);
}

// ------- cross attention: one wave per (b,h,qi), 256 keys, key mask --------
__global__ void cross_attn_k(const float* __restrict__ qproj, const float* __restrict__ kvproj,
                             const unsigned char* __restrict__ mask, bf16* __restrict__ cao) {
  int wid  = (int)((blockIdx.x * (size_t)blockDim.x + threadIdx.x) >> 5);
  int lane = threadIdx.x & 31;
  if (wid >= BQ * NH * NQ) return;
  int qi = wid & 31, h = (wid >> 5) & 7, b = wid >> 8;
  const float scale = 0.08838834764831845f;
  const float* qp = qproj + ((size_t)(b * NQ + qi)) * DD + h * HD + lane * 4;
  float q0 = qp[0], q1 = qp[1], q2 = qp[2], q3 = qp[3];
  float sarr[8];
#pragma unroll
  for (int jj = 0; jj < 8; ++jj) {
    float sl = -1e30f;
    for (int lj = 0; lj < 32; ++lj) {
      int j = jj * 32 + lj;
      const float* kp = kvproj + ((size_t)(b * LL + j)) * 2048 + h * HD + lane * 4;
      float p = q0 * kp[0] + q1 * kp[1] + q2 * kp[2] + q3 * kp[3];
      p = wave_sum(p) * scale;
      if (lane == lj) sl = (mask[b * LL + j] != 0) ? p : -1e30f;
    }
    sarr[jj] = sl;
  }
  float m = -1e30f;
#pragma unroll
  for (int jj = 0; jj < 8; ++jj) m = fmaxf(m, sarr[jj]);
  m = wave_max(m);
  float warr[8]; float ls = 0.0f;
#pragma unroll
  for (int jj = 0; jj < 8; ++jj) {
    warr[jj] = (sarr[jj] > -1e29f) ? __expf(sarr[jj] - m) : 0.0f;
    ls += warr[jj];
  }
  ls = wave_sum(ls);
  float o0 = 0, o1 = 0, o2 = 0, o3 = 0;
#pragma unroll
  for (int jj = 0; jj < 8; ++jj) {
    for (int lj = 0; lj < 32; ++lj) {
      int j = jj * 32 + lj;
      float w = __shfl(warr[jj], lj);
      const float* vp = kvproj + ((size_t)(b * LL + j)) * 2048 + 1024 + h * HD + lane * 4;
      o0 += w * vp[0]; o1 += w * vp[1]; o2 += w * vp[2]; o3 += w * vp[3];
    }
  }
  float inv = 1.0f / ls;
  bf16* op = cao + ((size_t)(b * NQ + qi)) * DD + h * HD + lane * 4;
  op[0] = (bf16)(o0 * inv); op[1] = (bf16)(o1 * inv);
  op[2] = (bf16)(o2 * inv); op[3] = (bf16)(o3 * inv);
}

// ---------------------------------------------------------------------------
extern "C" void kernel_launch(void* const* d_in, const int* in_sizes, int n_in,
                              void* d_out, int out_size, void* d_ws, size_t ws_size,
                              hipStream_t stream) {
  (void)in_sizes; (void)n_in; (void)out_size; (void)ws_size;
  // ---- inputs (setup_inputs order) ----
  const float* x        = (const float*)d_in[0];
  const float* text_tok = (const float*)d_in[1];
  const unsigned char* text_mask = (const unsigned char*)d_in[2];
  const float* conv1_w  = (const float*)d_in[3];
  const float* conv1_b  = (const float*)d_in[4];
  const float* conv2_w  = (const float*)d_in[5];
  const float* conv2_b  = (const float*)d_in[6];
  const float* query_t  = (const float*)d_in[7];
  const float* enc_in_w = (const float*)d_in[8];
  const float* enc_in_b = (const float*)d_in[9];
  const float* enc_out_w= (const float*)d_in[10];
  const float* enc_out_b= (const float*)d_in[11];
  const float* e_ln1_s  = (const float*)d_in[12];
  const float* e_ln1_b  = (const float*)d_in[13];
  const float* e_ln2_s  = (const float*)d_in[14];
  const float* e_ln2_b  = (const float*)d_in[15];
  const float* e_ff1_w  = (const float*)d_in[16];
  const float* e_ff1_b  = (const float*)d_in[17];
  const float* e_ff2_w  = (const float*)d_in[18];
  const float* e_ff2_b  = (const float*)d_in[19];
  const float* fin_ln_s = (const float*)d_in[20];
  const float* fin_ln_b = (const float*)d_in[21];
  const float* out_w    = (const float*)d_in[22];
  const float* out_b    = (const float*)d_in[23];
  const float* txt_w    = (const float*)d_in[24];
  const float* txt_b    = (const float*)d_in[25];
  const float* ca_in_w  = (const float*)d_in[26];
  const float* ca_in_b  = (const float*)d_in[27];
  const float* ca_out_w = (const float*)d_in[28];
  const float* ca_out_b = (const float*)d_in[29];
  const float* c_ln1_s  = (const float*)d_in[30];
  const float* c_ln1_b  = (const float*)d_in[31];
  const float* c_ln2_s  = (const float*)d_in[32];
  const float* c_ln2_b  = (const float*)d_in[33];
  const float* c_ff1_w  = (const float*)d_in[34];
  const float* c_ff1_b  = (const float*)d_in[35];
  const float* c_ff2_w  = (const float*)d_in[36];
  const float* c_ff2_b  = (const float*)d_in[37];

  float* qo = (float*)d_out;                       // [B*NQ, D] final/residual buffer

  // ---- workspace bump allocator ----
  char* wp = (char*)d_ws;
  auto alloc = [&](size_t bytes) -> void* {
    void* r = (void*)wp;
    wp += (bytes + 255) & ~(size_t)255;
    return r;
  };
  const size_t M_ENC = (size_t)BQ * SEQ;           // 8448
  bf16* c2w_b   = (bf16*)alloc((size_t)1024 * 3072 * 2);
  bf16* einw_b  = (bf16*)alloc((size_t)2 * 3072 * 1024 * 2);
  bf16* eoutw_b = (bf16*)alloc((size_t)2 * 1024 * 1024 * 2);
  bf16* ef1w_b  = (bf16*)alloc((size_t)2 * 4096 * 1024 * 2);
  bf16* ef2w_b  = (bf16*)alloc((size_t)2 * 1024 * 4096 * 2);
  bf16* outw_b  = (bf16*)alloc((size_t)1024 * 1024 * 2);
  bf16* txtw_b  = (bf16*)alloc((size_t)1024 * 768 * 2);
  bf16* cinw_b  = (bf16*)alloc((size_t)2 * 3072 * 1024 * 2);
  bf16* coutw_b = (bf16*)alloc((size_t)2 * 1024 * 1024 * 2);
  bf16* cf1w_b  = (bf16*)alloc((size_t)2 * 4096 * 1024 * 2);
  bf16* cf2w_b  = (bf16*)alloc((size_t)2 * 1024 * 4096 * 2);
  bf16* h1_b    = (bf16*)alloc((size_t)BQ * TT * DD * 2);
  bf16* A2_b    = (bf16*)alloc((size_t)BQ * TT * 3072 * 2);
  float* h2_f   = (float*)alloc((size_t)BQ * TT * DD * 4);
  float* hcat_f = (float*)alloc(M_ENC * DD * 4);
  bf16*  hn_b   = (bf16*)alloc(M_ENC * DD * 2);
  float* qkv_f  = (float*)alloc(M_ENC * 3072 * 4);
  bf16*  ao_b   = (bf16*)alloc(M_ENC * DD * 2);
  bf16*  mid_b  = (bf16*)alloc(M_ENC * 4096 * 2);
  bf16*  txt_b16= (bf16*)alloc((size_t)BQ * LL * TD * 2);
  float* tf_f   = (float*)alloc((size_t)BQ * LL * DD * 4);
  bf16*  xn_b   = (bf16*)alloc((size_t)BQ * NQ * DD * 2);
  bf16*  kvn_b  = (bf16*)alloc((size_t)BQ * LL * DD * 2);
  float* qproj_f= (float*)alloc((size_t)BQ * NQ * DD * 4);
  float* kvproj_f=(float*)alloc((size_t)BQ * LL * 2048 * 4);
  bf16*  cao_b  = (bf16*)alloc((size_t)BQ * NQ * DD * 2);
  bf16*  cmid_b = (bf16*)alloc((size_t)BQ * NQ * 4096 * 2);
  bf16*  qn_b   = (bf16*)alloc((size_t)BQ * NQ * DD * 2);

  auto cvt = [&](const float* src, bf16* dst, long n) {
    cvt_bf16_k<<<(unsigned)((n + 255) / 256), 256, 0, stream>>>(src, dst, n);
  };
  auto gemm = [&](const bf16* A, const bf16* W, const float* bias, const float* resid,
                  float* oF, bf16* oB, int M, int N, int K, int act) {
    dim3 g(N / 256, M / 16);
    gemm_wmma_k<<<g, 128, 0, stream>>>(A, W, bias, resid, oF, oB, M, N, K, act);
  };
  auto lnorm = [&](const float* xin, const float* g, const float* b, bf16* o,
                   int rows, int inner, int outer) {
    layernorm_bf16_k<<<rows, 256, 0, stream>>>(xin, g, b, o, inner, outer);
  };

  // ---- weight conversion (fp32 -> bf16, once) ----
  cvt(conv2_w,  c2w_b,   (long)1024 * 3072);
  cvt(enc_in_w, einw_b,  (long)2 * 3072 * 1024);
  cvt(enc_out_w,eoutw_b, (long)2 * 1024 * 1024);
  cvt(e_ff1_w,  ef1w_b,  (long)2 * 4096 * 1024);
  cvt(e_ff2_w,  ef2w_b,  (long)2 * 1024 * 4096);
  cvt(out_w,    outw_b,  (long)1024 * 1024);
  cvt(txt_w,    txtw_b,  (long)1024 * 768);
  cvt(ca_in_w,  cinw_b,  (long)2 * 3072 * 1024);
  cvt(ca_out_w, coutw_b, (long)2 * 1024 * 1024);
  cvt(c_ff1_w,  cf1w_b,  (long)2 * 4096 * 1024);
  cvt(c_ff2_w,  cf2w_b,  (long)2 * 1024 * 4096);
  cvt(text_tok, txt_b16, (long)BQ * LL * TD);

  // ---- visual frontend ----
  conv1_gelu_k<<<BQ * TT * 4, 256, 0, stream>>>(x, conv1_w, conv1_b, h1_b);
  im2col3_k<<<(unsigned)((size_t)BQ * TT * 3072 / 256), 256, 0, stream>>>(h1_b, A2_b);
  gemm(A2_b, c2w_b, conv2_b, nullptr, h2_f, nullptr, BQ * TT, DD, 3072, /*gelu*/1);
  build_seq_k<<<(unsigned)(M_ENC * DD / 256), 256, 0, stream>>>(h2_f, query_t, hcat_f);

  // ---- encoder layers (norm_first) ----
  for (int l = 0; l < 2; ++l) {
    lnorm(hcat_f, e_ln1_s + l * DD, e_ln1_b + l * DD, hn_b, (int)M_ENC, (int)M_ENC, 0);
    gemm(hn_b, einw_b + (size_t)l * 3072 * 1024, enc_in_b + l * 3072, nullptr,
         qkv_f, nullptr, (int)M_ENC, 3072, DD, 0);
    copy_v_seq_k<<<(unsigned)(M_ENC * DD / 256), 256, 0, stream>>>(qkv_f, ao_b);
    enc_qattn_k<<<BQ * NH * NQ / 8, 256, 0, stream>>>(qkv_f, ao_b);
    gemm(ao_b, eoutw_b + (size_t)l * 1024 * 1024, enc_out_b + l * DD, hcat_f,
         hcat_f, nullptr, (int)M_ENC, DD, DD, 0);
    lnorm(hcat_f, e_ln2_s + l * DD, e_ln2_b + l * DD, hn_b, (int)M_ENC, (int)M_ENC, 0);
    gemm(hn_b, ef1w_b + (size_t)l * 4096 * 1024, e_ff1_b + l * 4096, nullptr,
         nullptr, mid_b, (int)M_ENC, 4096, DD, /*gelu*/1);
    gemm(mid_b, ef2w_b + (size_t)l * 1024 * 4096, e_ff2_b + l * DD, hcat_f,
         hcat_f, nullptr, (int)M_ENC, DD, 4096, 0);
  }

  // ---- final LN on query rows + output projection -> qo (= d_out) ----
  lnorm(hcat_f, fin_ln_s, fin_ln_b, qn_b, BQ * NQ, NQ, SEQ);
  gemm(qn_b, outw_b, out_b, nullptr, qo, nullptr, BQ * NQ, DD, DD, 0);

  // ---- text projection ----
  gemm(txt_b16, txtw_b, txt_b, nullptr, tf_f, nullptr, BQ * LL, DD, TD, 0);

  // ---- cross-attention layers ----
  for (int l = 0; l < 2; ++l) {
    lnorm(qo, c_ln1_s + l * DD, c_ln1_b + l * DD, xn_b, BQ * NQ, BQ * NQ, 0);
    lnorm(tf_f, c_ln1_s + l * DD, c_ln1_b + l * DD, kvn_b, BQ * LL, BQ * LL, 0);
    gemm(xn_b, cinw_b + (size_t)l * 3072 * 1024, ca_in_b + l * 3072, nullptr,
         qproj_f, nullptr, BQ * NQ, DD, DD, 0);
    gemm(kvn_b, cinw_b + (size_t)l * 3072 * 1024 + (size_t)1024 * 1024,
         ca_in_b + l * 3072 + 1024, nullptr, kvproj_f, nullptr, BQ * LL, 2048, DD, 0);
    cross_attn_k<<<BQ * NH * NQ / 8, 256, 0, stream>>>(qproj_f, kvproj_f, text_mask, cao_b);
    gemm(cao_b, coutw_b + (size_t)l * 1024 * 1024, ca_out_b + l * DD, qo,
         qo, nullptr, BQ * NQ, DD, DD, 0);
    lnorm(qo, c_ln2_s + l * DD, c_ln2_b + l * DD, xn_b, BQ * NQ, BQ * NQ, 0);
    gemm(xn_b, cf1w_b + (size_t)l * 4096 * 1024, c_ff1_b + l * 4096, nullptr,
         nullptr, cmid_b, BQ * NQ, 4096, DD, /*gelu*/1);
    gemm(cmid_b, cf2w_b + (size_t)l * 1024 * 4096, c_ff2_b + l * DD, qo,
         qo, nullptr, BQ * NQ, DD, 4096, 0);
  }
}
